// LocalSelfAttention_21758304321690
// MI455X (gfx1250) — matrix operand order repaired and checked by reading, not compiled, not run on hardware
//
#include <hip/hip_runtime.h>

// ---------------------------------------------------------------------------
// Problem constants (from reference)
// ---------------------------------------------------------------------------
constexpr int Bc   = 4;
constexpr int Tc   = 4096;
constexpr int Cc   = 512;
constexpr int Hc   = 8;
constexpr int WINc = 15;
constexpr int PADc = WINc / 2;        // 7
constexpr int Dc   = Cc / Hc;         // 64
constexpr int TPc  = Tc + 2 * PADc;   // 4110 padded sequence
constexpr int Mp   = Bc * TPc;        // 16440 padded rows
constexpr int Mq   = Bc * Tc;         // 16384 real rows
constexpr float SCALEc = 2.70805020110221007f / 64.0f;  // log(15)/D

// ---------------------------------------------------------------------------
// Vector types for WMMA
// ---------------------------------------------------------------------------
typedef __attribute__((ext_vector_type(16))) __bf16 v16bf;
typedef __attribute__((ext_vector_type(8)))  __bf16 v8bf;
typedef __attribute__((ext_vector_type(8)))  float  v8f;

// ---------------------------------------------------------------------------
// Workspace layout (bytes)
// ---------------------------------------------------------------------------
constexpr size_t alignup(size_t x) { return (x + 255) & ~size_t(255); }
constexpr size_t OFF_XPAD = 0;
constexpr size_t SZ_XPAD  = (size_t)Mp * Cc * 2;                 // bf16 padded x
constexpr size_t OFF_WQT  = alignup(OFF_XPAD + SZ_XPAD);
constexpr size_t SZ_WQT   = (size_t)Cc * Cc * 2;                 // Wq^T bf16
constexpr size_t OFF_WKVT = alignup(OFF_WQT + SZ_WQT);
constexpr size_t SZ_WKVT  = (size_t)Cc * 2 * Cc * 2;             // Wkv^T bf16
constexpr size_t OFF_WPT  = alignup(OFF_WKVT + SZ_WKVT);
constexpr size_t SZ_WPT   = (size_t)Cc * Cc * 2;                 // Wproj^T bf16
constexpr size_t OFF_Q    = alignup(OFF_WPT + SZ_WPT);
constexpr size_t SZ_Q     = (size_t)Mp * Cc * 2;                 // q (padded rows) bf16
constexpr size_t OFF_KV   = alignup(OFF_Q + SZ_Q);
constexpr size_t SZ_KV    = (size_t)Mp * 2 * Cc * 2;             // kv (padded rows) bf16
constexpr size_t OFF_ATT  = alignup(OFF_KV + SZ_KV);
constexpr size_t SZ_ATT   = (size_t)Mq * Cc * 2;                 // attention out bf16

// ---------------------------------------------------------------------------
// CDNA5 async global->LDS copy (ASYNCcnt-tracked), inline asm.
// LDS destination address = low 32 bits of the generic pointer to a
// __shared__ object.
// ---------------------------------------------------------------------------
__device__ __forceinline__ void async_copy_b128(void* lds_dst, const void* gsrc)
{
    unsigned int ldst = (unsigned int)(size_t)lds_dst;
    asm volatile("global_load_async_to_lds_b128 %0, %1, off"
                 :: "v"(ldst), "v"(gsrc) : "memory");
}
__device__ __forceinline__ void wait_async_zero()
{
    asm volatile("s_wait_asynccnt 0" ::: "memory");
}

// ---------------------------------------------------------------------------
// Prep: masked + zero-padded x  ->  bf16 [B][TP][C]
// ---------------------------------------------------------------------------
__global__ __launch_bounds__(256) void prep_xpad_kernel(
    const float* __restrict__ x, const float* __restrict__ mask,
    __bf16* __restrict__ xpad)
{
    int idx = blockIdx.x * 256 + threadIdx.x;          // over Mp * C
    if (idx >= Mp * Cc) return;
    int c   = idx & (Cc - 1);
    int row = idx >> 9;                                // C == 512
    int b   = row / TPc;
    int r   = row - b * TPc;
    int t   = r - PADc;
    float v = 0.0f;
    if (t >= 0 && t < Tc) {
        int bt = b * Tc + t;
        v = x[(size_t)bt * Cc + c] * mask[bt];
    }
    xpad[idx] = (__bf16)v;
}

// ---------------------------------------------------------------------------
// Prep: W [K x N] fp32  ->  WT [N x K] bf16
// ---------------------------------------------------------------------------
__global__ __launch_bounds__(256) void transpose_w_kernel(
    const float* __restrict__ W, __bf16* __restrict__ WT, int K, int N)
{
    int idx = blockIdx.x * 256 + threadIdx.x;          // over N * K
    if (idx >= N * K) return;
    int n = idx / K;
    int k = idx - n * K;
    WT[idx] = (__bf16)W[(size_t)k * N + n];
}

// ---------------------------------------------------------------------------
// bf16 WMMA GEMM:  out[M,N] = A[M,K] * W[K,N] + bias[N]   (W given as WT[N,K])
// Tile: BM=128, BN=128, BK=32.  256 threads = 8 waves in a 4x2 grid; each
// wave owns a 32x64 sub-tile = 8 v8f accumulators, 8 WMMAs per K-step.
// Staging: double-buffered GLOBAL_LOAD_ASYNC_TO_LDS_B128 (ASYNCcnt), one
// barrier per K-step, loads of stage k+1 overlap WMMA of stage k.
// ---------------------------------------------------------------------------
template<bool OUT_F32, bool USE_MASK>
__global__ __launch_bounds__(256) void gemm_bf16_wmma_kernel(
    const __bf16* __restrict__ A, int lda,
    const __bf16* __restrict__ WT, int K,
    const float* __restrict__ bias,
    void* __restrict__ outp, int ldo,
    int M, int N, const float* __restrict__ maskrow)
{
    __shared__ __bf16 lA[2][128][40];   // 32 K-elems, pitch 40 (80B, 16B aligned)
    __shared__ __bf16 lB[2][128][40];

    const int tid  = threadIdx.x;
    const int wave = tid >> 5;
    const int lane = tid & 31;
    const int g    = lane >> 4;
    const int ln   = lane & 15;
    const int wr   = wave >> 1;         // wave row 0..3  -> rows wr*32..wr*32+31
    const int wc   = wave & 1;          // wave col 0..1  -> cols wc*64..wc*64+63

    const int m0 = blockIdx.y * 128;
    const int n0 = blockIdx.x * 128;

    v8f acc[2][4];
#pragma unroll
    for (int mt = 0; mt < 2; ++mt)
#pragma unroll
        for (int nt = 0; nt < 4; ++nt)
#pragma unroll
            for (int j = 0; j < 8; ++j) acc[mt][nt][j] = 0.0f;

    auto stage = [&](int buf, int k0) {
#pragma unroll
        for (int it = 0; it < 2; ++it) {
            int u    = tid + it * 256;
            int row  = u >> 2;
            int koff = (u & 3) * 8;
            int gr   = m0 + row;
            void* ldst = (void*)&lA[buf][row][koff];
            if (gr < M) {
                async_copy_b128(ldst, A + (size_t)gr * lda + k0 + koff);
            } else {
                *(uint4*)ldst = make_uint4(0u, 0u, 0u, 0u);
            }
        }
#pragma unroll
        for (int it = 0; it < 2; ++it) {
            int u    = tid + it * 256;
            int row  = u >> 2;
            int koff = (u & 3) * 8;
            async_copy_b128((void*)&lB[buf][row][koff],
                            WT + (size_t)(n0 + row) * K + k0 + koff);
        }
    };

    int buf = 0;
    stage(buf, 0);
    wait_async_zero();
    __syncthreads();

    for (int k0 = 0; k0 < K; k0 += 32) {
        if (k0 + 32 < K) stage(buf ^ 1, k0 + 32);

        v16bf afrag[2];
#pragma unroll
        for (int mt = 0; mt < 2; ++mt) {
            int mrow = wr * 32 + mt * 16 + ln;
            v8bf alo = *(const v8bf*)(&lA[buf][mrow][8 * g]);
            v8bf ahi = *(const v8bf*)(&lA[buf][mrow][16 + 8 * g]);
            afrag[mt] = __builtin_shufflevector(alo, ahi,
                0, 1, 2, 3, 4, 5, 6, 7, 8, 9, 10, 11, 12, 13, 14, 15);
        }
#pragma unroll
        for (int nt = 0; nt < 4; ++nt) {
            int nrow = wc * 64 + nt * 16 + ln;
            v8bf blo = *(const v8bf*)(&lB[buf][nrow][8 * g]);
            v8bf bhi = *(const v8bf*)(&lB[buf][nrow][16 + 8 * g]);
            v16bf bfrag = __builtin_shufflevector(blo, bhi,
                0, 1, 2, 3, 4, 5, 6, 7, 8, 9, 10, 11, 12, 13, 14, 15);
#pragma unroll
            for (int mt = 0; mt < 2; ++mt) {
                acc[mt][nt] = __builtin_amdgcn_wmma_f32_16x16x32_bf16(
                    false, afrag[mt], false, bfrag, (short)0, acc[mt][nt],
                    false, false);
            }
        }

        wait_async_zero();
        __syncthreads();
        buf ^= 1;
    }

#pragma unroll
    for (int mt = 0; mt < 2; ++mt) {
        int mbase = m0 + wr * 32 + mt * 16;
#pragma unroll
        for (int nt = 0; nt < 4; ++nt) {
            int gn = n0 + wc * 64 + nt * 16 + ln;
            float bi = bias[gn];
#pragma unroll
            for (int j = 0; j < 8; ++j) {
                int gm = mbase + j + 8 * g;
                if (gm < M) {
                    float v = acc[mt][nt][j] + bi;
                    if (USE_MASK) v *= maskrow[gm];
                    if (OUT_F32)
                        ((float*)outp)[(size_t)gm * ldo + gn] = v;
                    else
                        ((__bf16*)outp)[(size_t)gm * ldo + gn] = (__bf16)v;
                }
            }
        }
    }
}

// ---------------------------------------------------------------------------
// WMMA windowed attention. One wave per (b, h, 16-query tile).
//   S(16x32) = Q(16x64) x K^T       : B-fragments contiguous from global,
//                                     since B[d][key] walks d (K row-major).
//   softmax on banded 15-window     : lanes 0..15, wave-private LDS.
//   O(16x64) = P_band(16x32) x V    : V transposed into LDS (ds_store_b16
//                                     scatter), B-fragments from LDS.
// Out-of-range key columns carry P=0, so clamped edge reads are harmless.
// ---------------------------------------------------------------------------
__global__ __launch_bounds__(128) void attn_wmma_kernel(
    const __bf16* __restrict__ qbuf,   // [Mp][C]
    const __bf16* __restrict__ kvbuf,  // [Mp][2C]  (k | v)
    __bf16* __restrict__ abuf)         // [Mq][C]
{
    __shared__ float              sS [4][16][33];
    __shared__ __align__(16) __bf16 sP [4][16][40];
    __shared__ __align__(16) __bf16 sVT[4][64][40];

    const int tid  = threadIdx.x;
    const int wid  = tid >> 5;
    const int lane = tid & 31;
    const int g    = lane >> 4;
    const int ln   = lane & 15;

    // unit = ((b*H + h)*256 + tile)
    int unit = blockIdx.x * 4 + wid;
    int tile = unit & 255;
    int h    = (unit >> 8) & (Hc - 1);
    int b    = unit >> 11;
    int t0   = tile * 16;
    const size_t bTP = (size_t)b * TPc;

    // ---------------- Phase 1: scores S = Q x K^T ----------------
    v8f accS[2];
#pragma unroll
    for (int nt = 0; nt < 2; ++nt)
#pragma unroll
        for (int j = 0; j < 8; ++j) accS[nt][j] = 0.0f;

    const __bf16* qrow = qbuf + (bTP + PADc + t0 + ln) * Cc + h * Dc;
#pragma unroll
    for (int kd = 0; kd < 64; kd += 32) {
        v8bf alo = *(const v8bf*)(qrow + kd + 8 * g);
        v8bf ahi = *(const v8bf*)(qrow + kd + 16 + 8 * g);
        v16bf af = __builtin_shufflevector(alo, ahi,
            0, 1, 2, 3, 4, 5, 6, 7, 8, 9, 10, 11, 12, 13, 14, 15);
#pragma unroll
        for (int nt = 0; nt < 2; ++nt) {
            int krow = t0 + nt * 16 + ln;
            if (krow > TPc - 1) krow = TPc - 1;          // clamp (P=0 there)
            const __bf16* kr = kvbuf + (bTP + krow) * (2 * Cc) + h * Dc;
            v8bf blo = *(const v8bf*)(kr + kd + 8 * g);
            v8bf bhi = *(const v8bf*)(kr + kd + 16 + 8 * g);
            v16bf bf_ = __builtin_shufflevector(blo, bhi,
                0, 1, 2, 3, 4, 5, 6, 7, 8, 9, 10, 11, 12, 13, 14, 15);
            accS[nt] = __builtin_amdgcn_wmma_f32_16x16x32_bf16(
                false, af, false, bf_, (short)0, accS[nt], false, false);
        }
    }
    // spill S: C/D layout lane->(col n), vgpr j -> row m=j+8g
#pragma unroll
    for (int nt = 0; nt < 2; ++nt)
#pragma unroll
        for (int j = 0; j < 8; ++j)
            sS[wid][j + 8 * g][nt * 16 + ln] = accS[nt][j];

    // ---------------- Phase 2: banded softmax (lanes 0..15) ----------------
    if (g == 0) {
        int m = ln;
        float e[WINc];
        float mx = -3.4e38f;
#pragma unroll
        for (int w = 0; w < WINc; ++w) {
            e[w] = sS[wid][m][m + w] * SCALEc;
            mx = fmaxf(mx, e[w]);
        }
        float sum = 0.0f;
#pragma unroll
        for (int w = 0; w < WINc; ++w) { e[w] = __expf(e[w] - mx); sum += e[w]; }
        float inv = 1.0f / sum;
#pragma unroll
        for (int col = 0; col < 32; ++col)
            sP[wid][m][col] = (__bf16)0.0f;
#pragma unroll
        for (int w = 0; w < WINc; ++w)
            sP[wid][m][m + w] = (__bf16)(e[w] * inv);
    }

    // ---------------- Phase 3: O = P_band x V ----------------
    // Stage V^T: lane j owns key j; scatter 64 d-elements into sVT[d][j].
    {
        int vrow = t0 + lane;
        if (vrow > TPc - 1) vrow = TPc - 1;              // clamp (P=0 there)
        const __bf16* vp = kvbuf + (bTP + vrow) * (2 * Cc) + Cc + h * Dc;
        v8bf vv[8];
#pragma unroll
        for (int c = 0; c < 8; ++c) vv[c] = *(const v8bf*)(vp + c * 8);
#pragma unroll
        for (int c = 0; c < 8; ++c)
#pragma unroll
            for (int i = 0; i < 8; ++i)
                sVT[wid][c * 8 + i][lane] = vv[c][i];
    }

    // A fragment = P band row ln
    v8bf plo = *(const v8bf*)(&sP[wid][ln][8 * g]);
    v8bf phi = *(const v8bf*)(&sP[wid][ln][16 + 8 * g]);
    v16bf pf = __builtin_shufflevector(plo, phi,
        0, 1, 2, 3, 4, 5, 6, 7, 8, 9, 10, 11, 12, 13, 14, 15);

    v8f accO[4];
#pragma unroll
    for (int dt = 0; dt < 4; ++dt) {
#pragma unroll
        for (int j = 0; j < 8; ++j) accO[dt][j] = 0.0f;
        v8bf blo = *(const v8bf*)(&sVT[wid][dt * 16 + ln][8 * g]);
        v8bf bhi = *(const v8bf*)(&sVT[wid][dt * 16 + ln][16 + 8 * g]);
        v16bf bf_ = __builtin_shufflevector(blo, bhi,
            0, 1, 2, 3, 4, 5, 6, 7, 8, 9, 10, 11, 12, 13, 14, 15);
        accO[dt] = __builtin_amdgcn_wmma_f32_16x16x32_bf16(
            false, pf, false, bf_, (short)0, accO[dt], false, false);
    }

    // store: lane -> d column dt*16+ln, vgpr j -> query row t0+j+8g
#pragma unroll
    for (int dt = 0; dt < 4; ++dt)
#pragma unroll
        for (int j = 0; j < 8; ++j) {
            int t = t0 + j + 8 * g;
            abuf[(size_t)(b * Tc + t) * Cc + h * Dc + dt * 16 + ln] =
                (__bf16)accO[dt][j];
        }
}

// ---------------------------------------------------------------------------
// Launcher
// ---------------------------------------------------------------------------
extern "C" void kernel_launch(void* const* d_in, const int* in_sizes, int n_in,
                              void* d_out, int out_size, void* d_ws, size_t ws_size,
                              hipStream_t stream)
{
    const float* x     = (const float*)d_in[0];
    const float* mask  = (const float*)d_in[1];
    const float* Wq    = (const float*)d_in[2];
    const float* bq    = (const float*)d_in[3];
    const float* Wkv   = (const float*)d_in[4];
    const float* bkv   = (const float*)d_in[5];
    const float* Wproj = (const float*)d_in[6];
    const float* bproj = (const float*)d_in[7];

    char* ws = (char*)d_ws;
    __bf16* xpad  = (__bf16*)(ws + OFF_XPAD);
    __bf16* wqT   = (__bf16*)(ws + OFF_WQT);
    __bf16* wkvT  = (__bf16*)(ws + OFF_WKVT);
    __bf16* wpT   = (__bf16*)(ws + OFF_WPT);
    __bf16* qbuf  = (__bf16*)(ws + OFF_Q);
    __bf16* kvbuf = (__bf16*)(ws + OFF_KV);
    __bf16* abuf  = (__bf16*)(ws + OFF_ATT);

    // 1) masked + padded x -> bf16
    {
        int total = Mp * Cc;
        prep_xpad_kernel<<<(total + 255) / 256, 256, 0, stream>>>(x, mask, xpad);
    }
    // 2) transpose + convert weights
    transpose_w_kernel<<<(Cc * Cc + 255) / 256, 256, 0, stream>>>(Wq, wqT, Cc, Cc);
    transpose_w_kernel<<<(Cc * 2 * Cc + 255) / 256, 256, 0, stream>>>(Wkv, wkvT, Cc, 2 * Cc);
    transpose_w_kernel<<<(Cc * Cc + 255) / 256, 256, 0, stream>>>(Wproj, wpT, Cc, Cc);

    // 3) q = xpad @ Wq + bq   (computed on all padded rows; pads unused later)
    {
        dim3 grid(Cc / 128, (Mp + 127) / 128);
        gemm_bf16_wmma_kernel<false, false><<<grid, 256, 0, stream>>>(
            xpad, Cc, wqT, Cc, bq, (void*)qbuf, Cc, Mp, Cc, nullptr);
    }
    // 4) kv = xpad @ Wkv + bkv  (pad rows == bkv, matching reference pad semantics)
    {
        dim3 grid((2 * Cc) / 128, (Mp + 127) / 128);
        gemm_bf16_wmma_kernel<false, false><<<grid, 256, 0, stream>>>(
            xpad, Cc, wkvT, Cc, bkv, (void*)kvbuf, 2 * Cc, Mp, 2 * Cc, nullptr);
    }
    // 5) windowed softmax attention (WMMA): 8192 waves, 4 waves/block
    {
        attn_wmma_kernel<<<2048, 128, 0, stream>>>(qbuf, kvbuf, abuf);
    }
    // 6) out = (attn @ Wproj + bproj) * mask   -> fp32 d_out
    {
        dim3 grid(Cc / 128, (Mq + 127) / 128);
        gemm_bf16_wmma_kernel<true, true><<<grid, 256, 0, stream>>>(
            abuf, Cc, wpT, Cc, bproj, d_out, Cc, Mq, Cc, mask);
    }
    (void)in_sizes; (void)n_in; (void)out_size; (void)ws_size;
}
